// SpFisherLayer_30846455119900
// MI455X (gfx1250) — matrix operand order repaired and checked by reading, not compiled, not run on hardware
//
#include <hip/hip_runtime.h>

typedef __attribute__((ext_vector_type(16))) _Float16 v16h;
typedef __attribute__((ext_vector_type(8)))  float    v8f;

#define NROWS 32768
#define DDIM  256
#define KK    32
#define CHUNK 128
#define NBLK  128
#define CPB   2          // chunks per block: NBLK*CPB*CHUNK == NROWS
#define XS    136        // padded n-stride (halfs): 272B, 68 DW -> no 64-bank alias, 16B aligned
#define MS    264        // padded d-stride for mu (halfs): 528B, 132 DW
#define EPSF  1e-4f

__device__ __forceinline__ v16h ld2x8(const _Float16* pa, const _Float16* pb) {
  v16h r;
  ((uint4*)&r)[0] = *(const uint4*)pa;
  ((uint4*)&r)[1] = *(const uint4*)pb;
  return r;
}
__device__ __forceinline__ float rmax16(float v) {
#pragma unroll
  for (int m = 8; m >= 1; m >>= 1) v = fmaxf(v, __shfl_xor(v, m, 32));
  return v;
}
__device__ __forceinline__ float rsum16(float v) {
#pragma unroll
  for (int m = 8; m >= 1; m >>= 1) v += __shfl_xor(v, m, 32);
  return v;
}

// ---------------- Kernel 1: normalize mu, clip kappa, zero accumulators ----------------
__global__ __launch_bounds__(256) void spf_prep(
    const float* __restrict__ mu, const float* __restrict__ kappa,
    float* __restrict__ mu_n, _Float16* __restrict__ muh,
    float* __restrict__ kpos, float* __restrict__ skap,
    float* __restrict__ P, float* __restrict__ s_g, float* __restrict__ gk_g)
{
  int tid = threadIdx.x;
  int wv = tid >> 5, ln = tid & 31;
#pragma unroll
  for (int i = 0; i < 4; ++i) {
    int row = wv * 4 + i;
    const float* m = mu + row * DDIM;
    float vals[8];
    float ss = 0.f;
#pragma unroll
    for (int q = 0; q < 8; ++q) { vals[q] = m[ln + 32 * q]; ss += vals[q] * vals[q]; }
#pragma unroll
    for (int msk = 16; msk >= 1; msk >>= 1) ss += __shfl_xor(ss, msk, 32);
    float scale = 1.f / fmaxf(sqrtf(ss), EPSF);
#pragma unroll
    for (int q = 0; q < 8; ++q) {
      float v = vals[q] * scale;
      mu_n[row * DDIM + ln + 32 * q] = v;
      muh [row * DDIM + ln + 32 * q] = (_Float16)v;
    }
  }
  if (tid < KK) {
    float kp = fmaxf(kappa[tid], EPSF);
    kpos[tid] = kp;
    skap[tid] = sqrtf(kp);
    s_g[tid] = 0.f;
    gk_g[tid] = 0.f;
  }
  for (int i = tid; i < KK * DDIM; i += 256) P[i] = 0.f;
}

// ---------------- Kernel 2: main fused pass over x ----------------
__global__ __launch_bounds__(256) void spf_main(
    const float* __restrict__ x, const _Float16* __restrict__ muh,
    const float* __restrict__ kpos, const float* __restrict__ skap,
    float* __restrict__ P, float* __restrict__ s_g, float* __restrict__ gk_g)
{
  __shared__ _Float16 mu_lds[KK * MS];     // normalized mu, f16, padded rows
  __shared__ _Float16 xnT[DDIM * XS];      // raw x (f16), d-major transposed tile
  __shared__ _Float16 wT[KK * XS];         // scaled w, k-major transposed tile
  __shared__ float    s_acc[KK];
  __shared__ float    gk_acc[KK];

  const int tid = threadIdx.x;
  const int wv = tid >> 5;                 // wave id (0..7)
  const int ln = tid & 31;
  const int h  = ln >> 4;                  // lane half
  const int r  = ln & 15;

  // stage mu into LDS (u32 copies into padded rows)
  {
    const unsigned int* msrc = (const unsigned int*)muh;
    unsigned int* mdst = (unsigned int*)mu_lds;
    for (int i = tid; i < (KK * DDIM) / 2; i += 256) {
      int k = i >> 7, dp = i & 127;
      mdst[k * (MS / 2) + dp] = msrc[i];
    }
  }
  if (tid < KK) { s_acc[tid] = 0.f; gk_acc[tid] = 0.f; }

  const float kp0 = kpos[r], kp1 = kpos[r + 16];
  const float sk0 = skap[r], sk1 = skap[r + 16];

  v8f p00 = {}, p01 = {}, p10 = {}, p11 = {};

  for (int cc = 0; cc < CPB; ++cc) {
    const int chunk = blockIdx.x + cc * NBLK;
    const int g0 = chunk * CHUNK + wv * 16;          // wave's global row base
    const float* xrow = x + (size_t)(g0 + r) * DDIM; // lane's A-layout row
    const int n_local = wv * 16 + r;                 // chunk-local row index

    __syncthreads();  // prev phase-2 reads done; mu staged (iter 0)

    // --- phase A: load x, sumsq, build f16 A fragments, scatter raw f16 to xnT ---
    float ss = 0.f;
    v16h af[8];
#pragma unroll
    for (int s = 0; s < 8; ++s) {
      const float4* q0 = (const float4*)(xrow + 32 * s + 8 * h);
      const float4* q1 = (const float4*)(xrow + 32 * s + 16 + 8 * h);
      float4 c0 = q0[0], c1 = q0[1], c2 = q1[0], c3 = q1[1];
      ss += c0.x*c0.x + c0.y*c0.y + c0.z*c0.z + c0.w*c0.w
          + c1.x*c1.x + c1.y*c1.y + c1.z*c1.z + c1.w*c1.w
          + c2.x*c2.x + c2.y*c2.y + c2.z*c2.z + c2.w*c2.w
          + c3.x*c3.x + c3.y*c3.y + c3.z*c3.z + c3.w*c3.w;
      v16h a;
      a[0]=(_Float16)c0.x; a[1]=(_Float16)c0.y; a[2]=(_Float16)c0.z; a[3]=(_Float16)c0.w;
      a[4]=(_Float16)c1.x; a[5]=(_Float16)c1.y; a[6]=(_Float16)c1.z; a[7]=(_Float16)c1.w;
      a[8]=(_Float16)c2.x; a[9]=(_Float16)c2.y; a[10]=(_Float16)c2.z; a[11]=(_Float16)c2.w;
      a[12]=(_Float16)c3.x; a[13]=(_Float16)c3.y; a[14]=(_Float16)c3.z; a[15]=(_Float16)c3.w;
      af[s] = a;
      const int db0 = 32 * s + 8 * h;
      const int db1 = 32 * s + 16 + 8 * h;
#pragma unroll
      for (int e = 0; e < 8; ++e) xnT[(db0 + e) * XS + n_local] = a[e];
#pragma unroll
      for (int e = 0; e < 8; ++e) xnT[(db1 + e) * XS + n_local] = a[8 + e];
    }
    ss += __shfl_xor(ss, 16, 32);
    const float scale = 1.f / fmaxf(sqrtf(ss), EPSF);
    float rs[8];
#pragma unroll
    for (int j = 0; j < 8; ++j) rs[j] = __shfl(scale, j + 8 * h, 32);  // scale for C-row j+8h

    // --- c GEMM: c_raw = X_raw(f16) x mu_n^T, per-wave 16 rows x 32 clusters ---
    v8f cacc0 = {}, cacc1 = {};
#pragma unroll
    for (int s = 0; s < 8; ++s) {
      const _Float16* bp0 = &mu_lds[r * MS + 32 * s + 8 * h];
      const _Float16* bp1 = &mu_lds[(r + 16) * MS + 32 * s + 8 * h];
      v16h b0 = ld2x8(bp0, bp0 + 16);
      v16h b1 = ld2x8(bp1, bp1 + 16);
      cacc0 = __builtin_amdgcn_wmma_f32_16x16x32_f16(false, af[s], false, b0, (short)0, cacc0, false, false);
      cacc1 = __builtin_amdgcn_wmma_f32_16x16x32_f16(false, af[s], false, b1, (short)0, cacc1, false, false);
    }

    // --- softmax over k (per row), weights, G_kappa, s[k] ---
    float scp0 = 0.f, scp1 = 0.f, gkp0 = 0.f, gkp1 = 0.f;
#pragma unroll
    for (int j = 0; j < 8; ++j) {
      float cA = rs[j] * cacc0[j];          // true cosine: fold 1/||x|| back in
      float cB = rs[j] * cacc1[j];
      float vA = kp0 * cA, vB = kp1 * cB;
      float mx = rmax16(fmaxf(vA, vB));
      float eA = __expf(vA - mx), eB = __expf(vB - mx);
      float inv = 1.f / rsum16(eA + eB);
      float gA = eA * inv, gB = eB * inv;
      float dA = sqrtf(fmaxf(1.f - cA * cA, 0.f) + EPSF);
      float dB = sqrtf(fmaxf(1.f - cB * cB, 0.f) + EPSF);
      float wA = gA * sk0 / dA;
      float wB = gB * sk1 / dB;
      const int n = wv * 16 + j + 8 * h;    // C-fragment row -> chunk-local n
      wT[r * XS + n]        = (_Float16)(wA * rs[j]);   // scaled w pairs with raw-x tile
      wT[(r + 16) * XS + n] = (_Float16)(wB * rs[j]);
      scp0 += wA * cA;  scp1 += wB * cB;
      gkp0 += gA * 0.70710678118f * (kp0 * (1.f - cA) - 1.f);
      gkp1 += gB * 0.70710678118f * (kp1 * (1.f - cB) - 1.f);
    }
    scp0 += __shfl_xor(scp0, 16, 32);  scp1 += __shfl_xor(scp1, 16, 32);
    gkp0 += __shfl_xor(gkp0, 16, 32);  gkp1 += __shfl_xor(gkp1, 16, 32);
    if (h == 0) {
      atomicAdd(&s_acc[r], scp0);       atomicAdd(&s_acc[r + 16], scp1);
      atomicAdd(&gk_acc[r], gkp0);      atomicAdd(&gk_acc[r + 16], gkp1);
    }

    __syncthreads();  // xnT / wT complete for this chunk

    // --- phase 2: P_slice += W^T x X over this chunk; wave owns d-slice [32*wv, 32*wv+32) ---
    const int d0 = 32 * wv + r;
#pragma unroll
    for (int ns = 0; ns < 4; ++ns) {
      const int nb = 32 * ns;
      const _Float16* pa0 = &wT[r * XS + nb + 8 * h];
      const _Float16* pa1 = &wT[(r + 16) * XS + nb + 8 * h];
      v16h aw0 = ld2x8(pa0, pa0 + 16);
      v16h aw1 = ld2x8(pa1, pa1 + 16);
      const _Float16* pb0 = &xnT[d0 * XS + nb + 8 * h];
      const _Float16* pb1 = &xnT[(d0 + 16) * XS + nb + 8 * h];
      v16h bx0 = ld2x8(pb0, pb0 + 16);
      v16h bx1 = ld2x8(pb1, pb1 + 16);
      p00 = __builtin_amdgcn_wmma_f32_16x16x32_f16(false, aw0, false, bx0, (short)0, p00, false, false);
      p01 = __builtin_amdgcn_wmma_f32_16x16x32_f16(false, aw0, false, bx1, (short)0, p01, false, false);
      p10 = __builtin_amdgcn_wmma_f32_16x16x32_f16(false, aw1, false, bx0, (short)0, p10, false, false);
      p11 = __builtin_amdgcn_wmma_f32_16x16x32_f16(false, aw1, false, bx1, (short)0, p11, false, false);
    }
  }

  // --- flush P accumulators (C layout: row k = kt*16 + j + 8h, col d = 32*wv + 16*dt + r) ---
  {
    const int dc0 = 32 * wv + r, dc1 = dc0 + 16;
#pragma unroll
    for (int j = 0; j < 8; ++j) {
      const int k0 = j + 8 * h, k1 = 16 + j + 8 * h;
      atomicAdd(&P[k0 * DDIM + dc0], p00[j]);
      atomicAdd(&P[k0 * DDIM + dc1], p01[j]);
      atomicAdd(&P[k1 * DDIM + dc0], p10[j]);
      atomicAdd(&P[k1 * DDIM + dc1], p11[j]);
    }
  }
  __syncthreads();
  if (tid < KK) {
    atomicAdd(&s_g[tid],  s_acc[tid]);
    atomicAdd(&gk_g[tid], gk_acc[tid]);
  }
}

// ---------------- Kernel 3: finalize phi ----------------
__global__ __launch_bounds__(256) void spf_final(
    const float* __restrict__ P, const float* __restrict__ s_g,
    const float* __restrict__ gk_g, const float* __restrict__ mu_n,
    float* __restrict__ out)
{
  const int i = blockIdx.x * 256 + threadIdx.x;
  const float invN = 1.f / (float)NROWS;
  if (i < KK * DDIM) {
    const int k = i >> 8;
    out[i] = (P[i] - s_g[k] * mu_n[i]) * invN;
  } else if (i < KK * DDIM + KK) {
    out[i] = gk_g[i - KK * DDIM] * invN;
  }
}

extern "C" void kernel_launch(void* const* d_in, const int* in_sizes, int n_in,
                              void* d_out, int out_size, void* d_ws, size_t ws_size,
                              hipStream_t stream) {
  const float* x     = (const float*)d_in[0];   // 32768 x 256
  const float* mu    = (const float*)d_in[1];   // 32 x 256
  const float* kappa = (const float*)d_in[2];   // 32
  float* out = (float*)d_out;                   // 8224 f32

  // workspace carve
  float*     mu_n = (float*)d_ws;               // 8192 f32
  _Float16*  muh  = (_Float16*)(mu_n + KK * DDIM); // 8192 f16
  float*     kpos = (float*)(muh + KK * DDIM);  // 32
  float*     skap = kpos + KK;                  // 32
  float*     P    = skap + KK;                  // 8192 f32
  float*     s_g  = P + KK * DDIM;              // 32
  float*     gk_g = s_g + KK;                   // 32

  spf_prep <<<1,   256, 0, stream>>>(mu, kappa, mu_n, muh, kpos, skap, P, s_g, gk_g);
  spf_main <<<NBLK, 256, 0, stream>>>(x, muh, kpos, skap, P, s_g, gk_g);
  spf_final<<<(KK * DDIM + KK + 255) / 256, 256, 0, stream>>>(P, s_g, gk_g, mu_n, out);
}
